// mha1_13795434955004
// MI455X (gfx1250) — matrix operand order repaired and checked by reading, not compiled, not run on hardware
//
#include <hip/hip_runtime.h>

// ---- problem constants (from reference) ----
#define BATCH  2
#define SEQ    2048
#define DMODEL 1024
#define NH     16
#define DH     64      // DMODEL / NH
#define MROWS  (BATCH*SEQ)   // 4096

typedef __attribute__((ext_vector_type(16))) __bf16 v16bf;
typedef __attribute__((ext_vector_type(8)))  __bf16 v8bf;
typedef __attribute__((ext_vector_type(4)))  __bf16 v4bf;
typedef __attribute__((ext_vector_type(8)))  float  v8f;
typedef __attribute__((ext_vector_type(4)))  unsigned int v4u;
typedef __attribute__((ext_vector_type(8)))  int    v8i;
typedef __attribute__((ext_vector_type(4)))  int    v4i;

__device__ __forceinline__ v8f bfwmma(v16bf a, v16bf b, v8f c) {
  // D = A(16x32 bf16) * B(32x16 bf16) + C(16x16 f32)
  return __builtin_amdgcn_wmma_f32_16x16x32_bf16(false, a, false, b, (short)0, c,
                                                 false, false);
}

// A fragment (16x32, M x K), source row-major src[m][k]; base = tile origin.
// ISA layout: lanes 0-15 -> K 0..7,16..23 ; lanes 16-31 -> K 8..15,24..31
__device__ __forceinline__ v16bf load_a(const __bf16* base, int ld) {
  const int lane = threadIdx.x & 31;
  const int m  = lane & 15;
  const int kb = (lane < 16) ? 0 : 8;
  v8bf lo = *(const v8bf*)(base + m * ld + kb);
  v8bf hi = *(const v8bf*)(base + m * ld + kb + 16);
  return __builtin_shufflevector(lo, hi, 0,1,2,3,4,5,6,7,8,9,10,11,12,13,14,15);
}

// B fragment (32x16, K x N) where B(k,n) = src[n][k] (src holds N rows of K).
// Lanes 0-15: n=lane, K 0..7,16..23 ; lanes 16-31: n=lane-16, K 8..15,24..31.
__device__ __forceinline__ v16bf load_b(const __bf16* base, int ld) {
  const int lane = threadIdx.x & 31;
  const int n  = lane & 15;
  const int kb = (lane < 16) ? 0 : 8;
  v8bf lo = *(const v8bf*)(base + n * ld + kb);
  v8bf hi = *(const v8bf*)(base + n * ld + kb + 16);
  return __builtin_shufflevector(lo, hi, 0,1,2,3,4,5,6,7,8,9,10,11,12,13,14,15);
}

// ---- Tensor Data Mover: 2D tile (bf16) global -> LDS ----
// D# per CDNA5 ISA 8.3/8.4: group0 = {count, lds_addr, global_addr, type=2},
// group1 = {data_size=2B, tensor dims (large, no OOB), tile_w x tile_h,
//           tensor_dim0_stride}. Groups 2/3 zero (2-D tile).
// amdgpu-toolchain (clang-23) builtin: 6 args (g0, g1, g2, g3, g4, cpol).
__device__ __forceinline__ void tdm_load_2d(unsigned lds_addr, const void* gptr,
                                            unsigned tile_w, unsigned tile_h,
                                            unsigned stride_elems) {
  unsigned long long ga = (unsigned long long)gptr;
  v4u g0;
  g0[0] = 1u;                                       // count=1, user mode
  g0[1] = lds_addr;                                 // LDS byte address
  g0[2] = (unsigned)ga;                             // global_addr[31:0]
  g0[3] = (unsigned)((ga >> 32) & 0x1FFFFFFu) | (2u << 30);  // [56:32] | type=2
  v8i g1;
  g1[0] = (int)(1u << 16);                          // data_size=1 (2 bytes)
  g1[1] = 0;                                        // tensor_dim0 = 1<<20: lo16=0
  g1[2] = (int)0x10u;                               // dim0 hi16=0x10, dim1 lo16=0
  g1[3] = (int)(0x10u | (tile_w << 16));            // dim1 hi16=0x10 | tile_dim0
  g1[4] = (int)(tile_h & 0xFFFFu);                  // tile_dim1 (tile_dim2=0)
  g1[5] = (int)stride_elems;                        // tensor_dim0_stride[31:0]
  g1[6] = 0;                                        // stride hi / dim1_stride lo
  g1[7] = 0;
  v4i z4 = {0, 0, 0, 0};
  v8i z8 = {0, 0, 0, 0, 0, 0, 0, 0};
  __builtin_amdgcn_tensor_load_to_lds(g0, g1, z4, z4, z8, 0);
}

__device__ __forceinline__ unsigned lds_off(const void* p) {
  return (unsigned)(unsigned long long)p;   // low 32 bits of flat addr = LDS offset
}

// ---- fp32 -> bf16 conversion, 4-wide ----
__global__ void cvt4(const float* __restrict__ x, __bf16* __restrict__ y, int n4) {
  int i = blockIdx.x * blockDim.x + threadIdx.x;
  if (i < n4) {
    float4 f = ((const float4*)x)[i];
    v4bf o;
    o[0] = (__bf16)f.x; o[1] = (__bf16)f.y; o[2] = (__bf16)f.z; o[3] = (__bf16)f.w;
    ((v4bf*)y)[i] = o;
  }
}

// ---- projection GEMM with TDM-staged, double-buffered LDS tiles (K=64) ----
// P = X[4096x1024] @ W[1024x1024]^T + bias
// MODE 0: head-split row-major  kh/qh[((b*16+h)*SEQ + s)*DH + jj], c = jj*16 + h
// MODE 1: head-split transposed vt [((b*16+h)*DH + jj)*SEQ + s]
template<int MODE>
__global__ void proj_gemm(const __bf16* __restrict__ X, const __bf16* __restrict__ W,
                          const float* __restrict__ bias, __bf16* __restrict__ out) {
  __shared__ alignas(64) __bf16 Abuf[2][128 * 64];   // 32 KB
  __shared__ alignas(64) __bf16 Bbuf[2][64 * 64];    // 16 KB
  const int lane = threadIdx.x & 31;
  const int w    = threadIdx.x >> 5;       // 0..7
  const int wm   = w & 3, wn = w >> 2;
  const int mblk = blockIdx.x * 128;
  const int nblk = blockIdx.y * 64;
  const bool issuer = (w == 0);            // wave 0 drives the TDM

  if (issuer) {
    tdm_load_2d(lds_off(&Abuf[0][0]), X + (size_t)mblk * DMODEL, 64, 128, DMODEL);
    tdm_load_2d(lds_off(&Bbuf[0][0]), W + (size_t)nblk * DMODEL, 64, 64,  DMODEL);
  }

  v8f acc[2][2] = {};
  for (int k0 = 0; k0 < DMODEL; k0 += 64) {
    const int cur = (k0 >> 6) & 1;
    if (k0 + 64 < DMODEL) {
      if (issuer) {
        tdm_load_2d(lds_off(&Abuf[cur ^ 1][0]),
                    X + (size_t)mblk * DMODEL + (k0 + 64), 64, 128, DMODEL);
        tdm_load_2d(lds_off(&Bbuf[cur ^ 1][0]),
                    W + (size_t)nblk * DMODEL + (k0 + 64), 64, 64, DMODEL);
        __builtin_amdgcn_s_wait_tensorcnt(2);   // current buffer landed
      }
    } else if (issuer) {
      __builtin_amdgcn_s_wait_tensorcnt(0);
    }
    __syncthreads();                            // publish buffer `cur`

    const __bf16* At = &Abuf[cur][(wm * 32) * 64];
    const __bf16* Bt = &Bbuf[cur][(wn * 32) * 64];
    #pragma unroll
    for (int kk = 0; kk < 64; kk += 32) {
      v16bf a0 = load_a(At + kk,           64);
      v16bf a1 = load_a(At + 16 * 64 + kk, 64);
      v16bf b0 = load_b(Bt + kk,           64);
      v16bf b1 = load_b(Bt + 16 * 64 + kk, 64);
      acc[0][0] = bfwmma(a0, b0, acc[0][0]);
      acc[0][1] = bfwmma(a0, b1, acc[0][1]);
      acc[1][0] = bfwmma(a1, b0, acc[1][0]);
      acc[1][1] = bfwmma(a1, b1, acc[1][1]);
    }
    __syncthreads();                            // reads done before next DMA
  }

  const int hi = (lane >> 4) << 3;   // 0 or 8
  const int m0 = mblk + wm * 32;
  const int n0 = nblk + wn * 32;
  #pragma unroll
  for (int sm = 0; sm < 2; ++sm)
    #pragma unroll
    for (int sn = 0; sn < 2; ++sn) {
      const int n  = n0 + sn * 16 + (lane & 15);
      const float bv = bias[n];
      const int h  = n & 15, jj = n >> 4;
      #pragma unroll
      for (int r = 0; r < 8; ++r) {
        const int m = m0 + sm * 16 + r + hi;
        const int b = m >> 11, s = m & (SEQ - 1);
        const float v = acc[sm][sn][r] + bv;
        if (MODE == 0)
          out[((size_t)((b * NH + h) * SEQ + s)) * DH + jj] = (__bf16)v;
        else
          out[((size_t)((b * NH + h) * DH + jj)) * SEQ + s] = (__bf16)v;
      }
    }
}

// ---- fused attention: score + softmax-over-heads + attn @ V ----
// grid = BATCH * (SEQ/16) blocks, 512 threads (16 waves, wave h = head h)
__global__ void attn_kernel(const __bf16* __restrict__ kh, const __bf16* __restrict__ qh,
                            const __bf16* __restrict__ vt, __bf16* __restrict__ ao) {
  __shared__ alignas(64) float  scoref[32 * 16 * NH];   // [i_local][kcol][head] 32 KB
  __shared__ alignas(64) __bf16 attnT [NH * 16 * 32];   // [head][kcol][i_local] 16 KB

  const int lane = threadIdx.x & 31;
  const int h    = threadIdx.x >> 5;        // head = wave id, 0..15
  const int b    = blockIdx.x >> 7;         // SEQ/16 = 128 blocks per batch
  const int k0   = (blockIdx.x & 127) << 4; // query-column block

  const __bf16* khb = kh + ((size_t)(b * NH + h) * SEQ) * DH;
  const __bf16* qhb = qh + ((size_t)(b * NH + h) * SEQ) * DH;
  const __bf16* vtb = vt + ((size_t)(b * NH + h) * DH) * SEQ;

  // Q fragments are loop-invariant: B(k=j, n=kcol) = qh[k0+n][j]
  const v16bf bq0 = load_b(qhb + k0 * DH +  0, DH);
  const v16bf bq1 = load_b(qhb + k0 * DH + 32, DH);

  v8f acc[4] = {};
  const int n   = lane & 15;
  const int hi8 = (lane >> 4) << 3;

  for (int i0 = 0; i0 < SEQ; i0 += 32) {
    __builtin_prefetch(khb + (i0 + 32) * DH, 0, 3);
    #pragma unroll
    for (int half = 0; half < 2; ++half) {
      const int it = i0 + half * 16;
      v16bf a0 = load_a(khb + it * DH +  0, DH);
      v16bf a1 = load_a(khb + it * DH + 32, DH);
      v8f sc = {};
      sc = bfwmma(a0, bq0, sc);
      sc = bfwmma(a1, bq1, sc);
      #pragma unroll
      for (int r = 0; r < 8; ++r) {
        const int il = half * 16 + r + hi8;
        scoref[(il * 16 + n) * NH + h] = sc[r] * 0.125f;  // 1/sqrt(dh)
      }
    }
    __syncthreads();
    // softmax across the 16 heads; one (i_local, kcol) pair per thread
    {
      const int il = threadIdx.x >> 4;
      const int kc = threadIdx.x & 15;
      const float* p = &scoref[(il * 16 + kc) * NH];
      float mx = p[0];
      #pragma unroll
      for (int t = 1; t < NH; ++t) mx = fmaxf(mx, p[t]);
      float e[NH]; float s = 0.f;
      #pragma unroll
      for (int t = 0; t < NH; ++t) { e[t] = __expf(p[t] - mx); s += e[t]; }
      const float inv = 1.f / s;
      #pragma unroll
      for (int t = 0; t < NH; ++t)
        attnT[(t * 16 + kc) * 32 + il] = (__bf16)(e[t] * inv);
    }
    __syncthreads();
    // out[j,k] += sum_i v[i,j] * attn[i,k] ; A = vt (row-major [j][i]), B = attnT
    {
      const v16bf battn = load_b(&attnT[(h * 16) * 32], 32);
      #pragma unroll
      for (int jt = 0; jt < 4; ++jt) {
        v16bf av = load_a(vtb + (jt * 16) * SEQ + i0, SEQ);
        acc[jt] = bfwmma(av, battn, acc[jt]);
      }
    }
    __syncthreads();
  }

  // store re-interleaved: ao[b][s = k0+n][c = jj*16 + h]
  #pragma unroll
  for (int jt = 0; jt < 4; ++jt)
    #pragma unroll
    for (int r = 0; r < 8; ++r) {
      const int jj = jt * 16 + r + hi8;
      ao[((size_t)(b * SEQ + k0 + n)) * DMODEL + jj * NH + h] = (__bf16)acc[jt][r];
    }
}

// ---- output projection with TDM staging (K=64): res = AO @ Wo^T + bias ----
__global__ void out_gemm(const __bf16* __restrict__ A, const __bf16* __restrict__ W,
                         const float* __restrict__ bias, float* __restrict__ out) {
  __shared__ alignas(64) __bf16 Abuf[2][128 * 64];
  __shared__ alignas(64) __bf16 Bbuf[2][64 * 64];
  const int lane = threadIdx.x & 31;
  const int w    = threadIdx.x >> 5;
  const int wm   = w & 3, wn = w >> 2;
  const int mblk = blockIdx.x * 128;
  const int nblk = blockIdx.y * 64;
  const bool issuer = (w == 0);

  if (issuer) {
    tdm_load_2d(lds_off(&Abuf[0][0]), A + (size_t)mblk * DMODEL, 64, 128, DMODEL);
    tdm_load_2d(lds_off(&Bbuf[0][0]), W + (size_t)nblk * DMODEL, 64, 64,  DMODEL);
  }

  v8f acc[2][2] = {};
  for (int k0 = 0; k0 < DMODEL; k0 += 64) {
    const int cur = (k0 >> 6) & 1;
    if (k0 + 64 < DMODEL) {
      if (issuer) {
        tdm_load_2d(lds_off(&Abuf[cur ^ 1][0]),
                    A + (size_t)mblk * DMODEL + (k0 + 64), 64, 128, DMODEL);
        tdm_load_2d(lds_off(&Bbuf[cur ^ 1][0]),
                    W + (size_t)nblk * DMODEL + (k0 + 64), 64, 64, DMODEL);
        __builtin_amdgcn_s_wait_tensorcnt(2);
      }
    } else if (issuer) {
      __builtin_amdgcn_s_wait_tensorcnt(0);
    }
    __syncthreads();

    const __bf16* At = &Abuf[cur][(wm * 32) * 64];
    const __bf16* Bt = &Bbuf[cur][(wn * 32) * 64];
    #pragma unroll
    for (int kk = 0; kk < 64; kk += 32) {
      v16bf a0 = load_a(At + kk,           64);
      v16bf a1 = load_a(At + 16 * 64 + kk, 64);
      v16bf b0 = load_b(Bt + kk,           64);
      v16bf b1 = load_b(Bt + 16 * 64 + kk, 64);
      acc[0][0] = bfwmma(a0, b0, acc[0][0]);
      acc[0][1] = bfwmma(a0, b1, acc[0][1]);
      acc[1][0] = bfwmma(a1, b0, acc[1][0]);
      acc[1][1] = bfwmma(a1, b1, acc[1][1]);
    }
    __syncthreads();
  }

  const int hi = (lane >> 4) << 3;
  const int m0 = mblk + wm * 32;
  const int n0 = nblk + wn * 32;
  #pragma unroll
  for (int sm = 0; sm < 2; ++sm)
    #pragma unroll
    for (int sn = 0; sn < 2; ++sn) {
      const int nn = n0 + sn * 16 + (lane & 15);
      const float bv = bias[nn];
      #pragma unroll
      for (int r = 0; r < 8; ++r)
        out[((size_t)(m0 + sm * 16 + r + hi)) * DMODEL + nn] = acc[sm][sn][r] + bv;
    }
}

extern "C" void kernel_launch(void* const* d_in, const int* in_sizes, int n_in,
                              void* d_out, int out_size, void* d_ws, size_t ws_size,
                              hipStream_t stream) {
  const float* KEY   = (const float*)d_in[0];
  const float* VALUE = (const float*)d_in[1];
  const float* QUERY = (const float*)d_in[2];
  const float* Wk_w  = (const float*)d_in[3];
  const float* Wk_b  = (const float*)d_in[4];
  const float* Wq_w  = (const float*)d_in[5];
  const float* Wq_b  = (const float*)d_in[6];
  const float* Wv_w  = (const float*)d_in[7];
  const float* Wv_b  = (const float*)d_in[8];
  const float* Wo_w  = (const float*)d_in[9];
  const float* Wo_b  = (const float*)d_in[10];

  const size_t szX = (size_t)BATCH * SEQ * DMODEL;   // 4,194,304
  const size_t szW = (size_t)DMODEL * DMODEL;        // 1,048,576

  __bf16* p    = (__bf16*)d_ws;
  __bf16* Kbf  = p; p += szX;
  __bf16* Qbf  = p; p += szX;
  __bf16* Vbf  = p; p += szX;
  __bf16* Wkbf = p; p += szW;
  __bf16* Wqbf = p; p += szW;
  __bf16* Wvbf = p; p += szW;
  __bf16* Wobf = p; p += szW;
  __bf16* khp  = p; p += szX;
  __bf16* qhp  = p; p += szX;
  __bf16* vtp  = p; p += szX;
  __bf16* aop  = p; p += szX;   // total 32M bf16 = 64 MiB

  const int gX = (int)(szX / 4 + 255) / 256;
  const int gW = (int)(szW / 4 + 255) / 256;
  cvt4<<<gX, 256, 0, stream>>>(KEY,   Kbf,  (int)(szX / 4));
  cvt4<<<gX, 256, 0, stream>>>(QUERY, Qbf,  (int)(szX / 4));
  cvt4<<<gX, 256, 0, stream>>>(VALUE, Vbf,  (int)(szX / 4));
  cvt4<<<gW, 256, 0, stream>>>(Wk_w,  Wkbf, (int)(szW / 4));
  cvt4<<<gW, 256, 0, stream>>>(Wq_w,  Wqbf, (int)(szW / 4));
  cvt4<<<gW, 256, 0, stream>>>(Wv_w,  Wvbf, (int)(szW / 4));
  cvt4<<<gW, 256, 0, stream>>>(Wo_w,  Wobf, (int)(szW / 4));

  dim3 gproj(MROWS / 128, DMODEL / 64);
  proj_gemm<0><<<gproj, 256, 0, stream>>>(Kbf, Wkbf, Wk_b, khp);
  proj_gemm<0><<<gproj, 256, 0, stream>>>(Qbf, Wqbf, Wq_b, qhp);
  proj_gemm<1><<<gproj, 256, 0, stream>>>(Vbf, Wvbf, Wv_b, vtp);

  attn_kernel<<<BATCH * (SEQ / 16), 512, 0, stream>>>(khp, qhp, vtp, aop);

  out_gemm<<<gproj, 256, 0, stream>>>(aop, Wobf, Wo_b, (float*)d_out);
}
